// LSTMTagger_10350871183551
// MI455X (gfx1250) — compile-verified, matrix operand order checked by reading, not run on hardware
//
#include <hip/hip_runtime.h>
#include <hip/hip_bf16.h>

// Problem constants (from reference)
#define SEQ   8192
#define LCH   12
#define WEv   128
#define CEv   64
#define CRv   64
#define HDIM  256
#define TOUT  128
#define ZC    256    // 4*CR
#define ZW    1024   // 4*H

// WMMA tiling
#define NT1 16       // char: 256 out feats / 16
#define KC1 4        // char: K=128 / 32
#define NT2 64       // word in-proj: 1024 / 16
#define KC2 6        // word in-proj: K=192 / 32
#define NT4 8        // out proj: 128 / 16
#define KC4 8        // out proj: K=256 / 32

typedef __attribute__((ext_vector_type(16))) __bf16        v16bf;
typedef __attribute__((ext_vector_type(2)))  __bf16        v2bf;
typedef __attribute__((ext_vector_type(8)))  float         v8f;
typedef __attribute__((ext_vector_type(4)))  unsigned int  v4u;

union FragU { unsigned int u[8]; v4u q[2]; v16bf v; };
union PairU { unsigned int u; v2bf v; };

static __device__ __forceinline__ unsigned int bfpack2(float a, float b) {
#if __has_builtin(__builtin_amdgcn_cvt_pk_bf16_f32)
  PairU p; p.v = __builtin_amdgcn_cvt_pk_bf16_f32(a, b);
  return p.u;
#else
  unsigned int x = __float_as_uint(a); x = (x + 0x7FFFu + ((x >> 16) & 1u)) >> 16;
  unsigned int y = __float_as_uint(b); y = (y + 0x7FFFu + ((y >> 16) & 1u)) >> 16;
  return (x & 0xFFFFu) | (y << 16);
#endif
}

static __device__ __forceinline__ float fast_rcp(float x) {
#if __has_builtin(__builtin_amdgcn_rcpf)
  return __builtin_amdgcn_rcpf(x);
#else
  return 1.0f / x;
#endif
}
// Branch-free gate nonlinearities (v_exp + v_rcp, no libm branches):
static __device__ __forceinline__ float sigm(float x) {
  return fast_rcp(1.0f + __expf(-x));
}
static __device__ __forceinline__ float tanh_fast(float x) {
  // tanh(x) = 1 - 2/(exp(2x)+1); saturates correctly at +-1
  return 1.0f - 2.0f * fast_rcp(__expf(2.0f * x) + 1.0f);
}
// ISA 16-bit A/B fragment K index for VGPR v within a 32-K chunk
static __device__ __forceinline__ int kpair(int base, int v) {
  return base + (v < 4 ? 2 * v : 16 + 2 * (v - 4));
}

// Opaque LDS fragment load: two ds_load_b128 from a 32-bit LDS offset.
// asm volatile => no LICM hoist, no register caching, no spills.
static __device__ __forceinline__ v16bf lds_frag_load(unsigned lds_byte_addr) {
  FragU a;
  asm volatile("ds_load_b128 %0, %2\n\t"
               "ds_load_b128 %1, %2 offset:16\n\t"
               "s_wait_dscnt 0"
               : "=v"(a.q[0]), "=v"(a.q[1])
               : "v"(lds_byte_addr)
               : "memory");
  return a.v;
}

// ---------------------------------------------------------------------------
// Prep: pack weights into bf16 WMMA A-fragment layout; combine biases.
// Fragment buffer layout: uint index = ((nt*KC + kc)*32 + lane)*8 + v
// ---------------------------------------------------------------------------
__global__ void prep_kernel(const float* __restrict__ c_wih, const float* __restrict__ c_whh,
                            const float* __restrict__ c_bih, const float* __restrict__ c_bhh,
                            const float* __restrict__ w_wih,
                            const float* __restrict__ w_bih, const float* __restrict__ w_bhh,
                            const float* __restrict__ out_w,
                            unsigned int* __restrict__ wcf, unsigned int* __restrict__ wif,
                            unsigned int* __restrict__ owf,
                            float* __restrict__ cb, float* __restrict__ wb) {
  const int NWC = NT1 * KC1 * 32 * 8;
  const int NWI = NT2 * KC2 * 32 * 8;
  const int NOW = NT4 * KC4 * 32 * 8;
  int idx = blockIdx.x * blockDim.x + threadIdx.x;

  if (idx < NWC) {
    int v = idx & 7, lane = (idx >> 3) & 31, fi = idx >> 8;
    int kc = fi % KC1, nt = fi / KC1;
    int m = nt * 16 + (lane & 15);
    int base = (lane < 16) ? 0 : 8;
    int k0 = kc * 32 + kpair(base, v);
    float f0, f1;
    if (k0 < CEv) { f0 = c_wih[m * CEv + k0];         f1 = c_wih[m * CEv + k0 + 1]; }
    else          { f0 = c_whh[m * CRv + (k0 - CEv)]; f1 = c_whh[m * CRv + (k0 - CEv) + 1]; }
    wcf[idx] = bfpack2(f0, f1);
    return;
  }
  idx -= NWC;
  if (idx < NWI) {
    int v = idx & 7, lane = (idx >> 3) & 31, fi = idx >> 8;
    int kc = fi % KC2, nt = fi / KC2;
    int m = nt * 16 + (lane & 15);
    int base = (lane < 16) ? 0 : 8;
    int k0 = kc * 32 + kpair(base, v);
    wif[idx] = bfpack2(w_wih[m * 192 + k0], w_wih[m * 192 + k0 + 1]);
    return;
  }
  idx -= NWI;
  if (idx < NOW) {
    int v = idx & 7, lane = (idx >> 3) & 31, fi = idx >> 8;
    int kc = fi % KC4, nt = fi / KC4;
    int m = nt * 16 + (lane & 15);
    int base = (lane < 16) ? 0 : 8;
    int k0 = kc * 32 + kpair(base, v);
    owf[idx] = bfpack2(out_w[m * HDIM + k0], out_w[m * HDIM + k0 + 1]);
    return;
  }
  idx -= NOW;
  if (idx < ZC) { cb[idx] = c_bih[idx] + c_bhh[idx]; return; }
  idx -= ZC;
  if (idx < ZW) { wb[idx] = w_bih[idx] + w_bhh[idx]; return; }
}

// ---------------------------------------------------------------------------
// Phase 1: char LSTM. One wave owns 16 words. z^T = Wc @ [x;h]^T via WMMA.
// - Wc fragments (64 KB) staged into LDS once per block via
//   global_load_async_to_lds_b128 (ASYNCcnt path).
// - A-fragments served per-use by opaque ds_load_b128 asm (no hoist/spill).
// - N-tiles processed in ti-groups {i,f,g,o} so only 4 v8f accs are live.
// ---------------------------------------------------------------------------
__global__ __launch_bounds__(128) __attribute__((amdgpu_waves_per_eu(4))) void
char_lstm_kernel(const int* __restrict__ char_ixs, const int* __restrict__ char_lens,
                 const float* __restrict__ char_emb,
                 const unsigned int* __restrict__ wcf, const float* __restrict__ cb,
                 float* __restrict__ h_char) {
  __shared__ unsigned int ldsWcf[NT1 * KC1 * 32 * 8];   // 16384 uints = 64 KB
  // Low 32 bits of the generic shared address == LDS byte offset (ISA aperture)
  unsigned lbase = (unsigned)(uintptr_t)ldsWcf;

  // --- async stage: 4096 chunks of 16B, 128 threads x 32 chunks each ---
  {
#pragma unroll 4
    for (int i = 0; i < 32; ++i) {
      int c = i * 128 + (int)threadIdx.x;
      unsigned long long ga = (unsigned long long)(uintptr_t)(wcf + (size_t)c * 4);
      unsigned la = lbase + (unsigned)c * 16u;
      asm volatile("global_load_async_to_lds_b128 %0, %1, off"
                   :: "v"(la), "v"(ga) : "memory");
    }
    asm volatile("s_wait_asynccnt 0" ::: "memory");
  }
  __syncthreads();

  int tid  = blockIdx.x * blockDim.x + threadIdx.x;
  int lane = tid & 31;
  int wave = tid >> 5;
  if (wave >= SEQ / 16) return;
  int base = (lane < 16) ? 0 : 8;
  int w    = wave * 16 + (lane & 15);
  int len  = char_lens[w];
  unsigned laneOff = lbase + (unsigned)lane * 32u;      // + frag*1024

  float cst[4][8], hval[4][8];
  FragU hf[2];
#pragma unroll
  for (int ti = 0; ti < 4; ++ti)
#pragma unroll
    for (int r = 0; r < 8; ++r) { cst[ti][r] = 0.f; hval[ti][r] = 0.f; }
#pragma unroll
  for (int c = 0; c < 2; ++c)
#pragma unroll
    for (int v = 0; v < 8; ++v) hf[c].u[v] = 0u;

  const v8f vzero = {0.f, 0.f, 0.f, 0.f, 0.f, 0.f, 0.f, 0.f};

#pragma unroll 1
  for (int t = 0; t < LCH; ++t) {
    int cix = char_ixs[w * LCH + t];
    const float* xe = char_emb + (size_t)cix * CEv;
    FragU xf[2];
#pragma unroll
    for (int c = 0; c < 2; ++c)
#pragma unroll
      for (int v = 0; v < 8; ++v) {
        int k0 = c * 32 + kpair(base, v);
        xf[c].u[v] = bfpack2(xe[k0], xe[k0 + 1]);
      }

    bool m = (t < len);
#pragma unroll
    for (int ti = 0; ti < 4; ++ti) {
      v8f acc4[4];
#pragma unroll
      for (int g = 0; g < 4; ++g) acc4[g] = vzero;
#pragma unroll
      for (int g = 0; g < 4; ++g) {
        int nt = g * 4 + ti;                     // i/f/g/o tile for slice ti
#pragma unroll
        for (int kc = 0; kc < KC1; ++kc) {
          v16bf a = lds_frag_load(laneOff + (unsigned)(nt * KC1 + kc) * 1024u);
          v16bf b = (kc < 2) ? xf[kc].v : hf[kc - 2].v;
          acc4[g] = __builtin_amdgcn_wmma_f32_16x16x32_bf16(false, a, false, b,
                                                            (short)0, acc4[g], false, false);
        }
      }
#pragma unroll
      for (int r = 0; r < 8; ++r) {
        int j = 16 * ti + base + r;              // gate-local feature
        float zi = acc4[0][r] + cb[j];
        float zf = acc4[1][r] + cb[64 + j];
        float zg = acc4[2][r] + cb[128 + j];
        float zo = acc4[3][r] + cb[192 + j];
        float c2 = sigm(zf) * cst[ti][r] + sigm(zi) * tanh_fast(zg);
        float h2 = sigm(zo) * tanh_fast(c2);
        cst[ti][r]  = m ? c2 : cst[ti][r];
        hval[ti][r] = m ? h2 : hval[ti][r];
      }
    }
    // Rebuild h as B fragments for K-chunks 2,3 (entirely lane-local)
#pragma unroll
    for (int c = 0; c < 2; ++c)
#pragma unroll
      for (int v = 0; v < 8; ++v) {
        int tt = 2 * c + (v >= 4 ? 1 : 0);
        int r0 = 2 * (v & 3);
        hf[c].u[v] = bfpack2(hval[tt][r0], hval[tt][r0 + 1]);
      }
  }
#pragma unroll
  for (int ti = 0; ti < 4; ++ti)
#pragma unroll
    for (int r = 0; r < 8; ++r)
      h_char[(size_t)w * CRv + 16 * ti + base + r] = hval[ti][r];
}

// ---------------------------------------------------------------------------
// Phase 2: hoisted word-LSTM input projection  XZ = [wemb, h_char] @ Wih^T + wb
// ---------------------------------------------------------------------------
__global__ __launch_bounds__(128) __attribute__((amdgpu_waves_per_eu(4))) void
word_inproj_kernel(const int* __restrict__ word_ixs, const float* __restrict__ word_emb,
                   const float* __restrict__ h_char,
                   const unsigned int* __restrict__ wif, const float* __restrict__ wb,
                   float* __restrict__ XZ) {
  int tid  = blockIdx.x * blockDim.x + threadIdx.x;
  int lane = tid & 31;
  int wave = tid >> 5;
  if (wave >= SEQ / 16) return;
  int base = (lane < 16) ? 0 : 8;
  int w    = wave * 16 + (lane & 15);
  int wix  = word_ixs[w];
  const float* we = word_emb + (size_t)wix * WEv;
  const float* hc = h_char + (size_t)w * CRv;

  FragU bf[KC2];
#pragma unroll
  for (int c = 0; c < KC2; ++c)
#pragma unroll
    for (int v = 0; v < 8; ++v) {
      int k0 = c * 32 + kpair(base, v);
      float f0 = (k0 < WEv) ? we[k0] : hc[k0 - WEv];
      float f1 = (k0 + 1 < WEv) ? we[k0 + 1] : hc[k0 + 1 - WEv];
      bf[c].u[v] = bfpack2(f0, f1);
    }

  const v8f vzero = {0.f, 0.f, 0.f, 0.f, 0.f, 0.f, 0.f, 0.f};
#pragma unroll 1
  for (int nt = 0; nt < NT2; ++nt) {          // rolled: keep VGPRs low
    v8f acc = vzero;
#pragma unroll
    for (int kc = 0; kc < KC2; ++kc) {
      v16bf a = *reinterpret_cast<const v16bf*>(wif + (size_t)((nt * KC2 + kc) * 32 + lane) * 8);
      acc = __builtin_amdgcn_wmma_f32_16x16x32_bf16(false, a, false, bf[kc].v,
                                                    (short)0, acc, false, false);
    }
#pragma unroll
    for (int r = 0; r < 8; ++r) {
      int feat = nt * 16 + base + r;
      XZ[(size_t)w * ZW + feat] = acc[r] + wb[feat];
    }
  }
}

// ---------------------------------------------------------------------------
// Phase 3: serial word LSTM. Single persistent workgroup; only the 256->1024
// recurrent GEMV + gates remain on the critical path. w_whh stays L2-hot.
// ---------------------------------------------------------------------------
__global__ __launch_bounds__(1024, 1) void
word_lstm_kernel(const float* __restrict__ XZ, const float* __restrict__ w_whh,
                 float* __restrict__ hs) {
  __shared__ float hsh[HDIM];
  __shared__ float zsh[ZW];
  int j = threadIdx.x;
  if (j < HDIM) hsh[j] = 0.f;
  float c = 0.f;
  __syncthreads();

  const float4* wrow = (const float4*)(w_whh + (size_t)j * HDIM);
#pragma unroll 1
  for (int t = 0; t < SEQ; ++t) {
    float acc = XZ[(size_t)t * ZW + j];
    if (t + 1 < SEQ) __builtin_prefetch(XZ + (size_t)(t + 1) * ZW + j, 0, 0);
    const float4* h4 = (const float4*)hsh;
#pragma unroll 8
    for (int k = 0; k < HDIM / 4; ++k) {
      float4 wv = wrow[k];
      float4 hv = h4[k];
      acc = fmaf(wv.x, hv.x, acc);
      acc = fmaf(wv.y, hv.y, acc);
      acc = fmaf(wv.z, hv.z, acc);
      acc = fmaf(wv.w, hv.w, acc);
    }
    zsh[j] = acc;
    __syncthreads();
    if (j < HDIM) {
      float zi = zsh[j], zf = zsh[HDIM + j], zg = zsh[2 * HDIM + j], zo = zsh[3 * HDIM + j];
      float c2 = sigm(zf) * c + sigm(zi) * tanh_fast(zg);
      c = c2;
      float hv = sigm(zo) * tanh_fast(c2);
      hsh[j] = hv;
      hs[(size_t)t * HDIM + j] = hv;
    }
    __syncthreads();
  }
}

// ---------------------------------------------------------------------------
// Phase 4: logits = hs @ out_w^T + b, then log_softmax over 128 tags.
// One accumulator tile live at a time; each word's logits live in lanes
// l and l^16 -> shfl_xor(16) reduction for max / sum-exp.
// ---------------------------------------------------------------------------
__global__ __launch_bounds__(128) __attribute__((amdgpu_waves_per_eu(4))) void
out_softmax_kernel(const float* __restrict__ hs, const unsigned int* __restrict__ owf,
                   const float* __restrict__ out_b, float* __restrict__ out) {
  int tid  = blockIdx.x * blockDim.x + threadIdx.x;
  int lane = tid & 31;
  int wave = tid >> 5;
  if (wave >= SEQ / 16) return;
  int base = (lane < 16) ? 0 : 8;
  int w    = wave * 16 + (lane & 15);
  const float* hrow = hs + (size_t)w * HDIM;

  FragU bf[KC4];
#pragma unroll
  for (int c = 0; c < KC4; ++c)
#pragma unroll
    for (int v = 0; v < 8; ++v) {
      int k0 = c * 32 + kpair(base, v);
      bf[c].u[v] = bfpack2(hrow[k0], hrow[k0 + 1]);
    }

  const v8f vzero = {0.f, 0.f, 0.f, 0.f, 0.f, 0.f, 0.f, 0.f};
  float zv[NT4][8];
#pragma unroll
  for (int nt = 0; nt < NT4; ++nt) {
    v8f acc = vzero;
#pragma unroll
    for (int kc = 0; kc < KC4; ++kc) {
      v16bf a = *reinterpret_cast<const v16bf*>(owf + (size_t)((nt * KC4 + kc) * 32 + lane) * 8);
      acc = __builtin_amdgcn_wmma_f32_16x16x32_bf16(false, a, false, bf[kc].v,
                                                    (short)0, acc, false, false);
    }
#pragma unroll
    for (int r = 0; r < 8; ++r)
      zv[nt][r] = acc[r] + out_b[nt * 16 + base + r];
  }

  float mx = -3.0e38f;
#pragma unroll
  for (int nt = 0; nt < NT4; ++nt)
#pragma unroll
    for (int r = 0; r < 8; ++r) mx = fmaxf(mx, zv[nt][r]);
  mx = fmaxf(mx, __shfl_xor(mx, 16, 32));
  float sum = 0.f;
#pragma unroll
  for (int nt = 0; nt < NT4; ++nt)
#pragma unroll
    for (int r = 0; r < 8; ++r) sum += __expf(zv[nt][r] - mx);
  sum += __shfl_xor(sum, 16, 32);
  float lse = __logf(sum);
#pragma unroll
  for (int nt = 0; nt < NT4; ++nt)
#pragma unroll
    for (int r = 0; r < 8; ++r)
      out[(size_t)w * TOUT + nt * 16 + base + r] = zv[nt][r] - mx - lse;
}

// ---------------------------------------------------------------------------
extern "C" void kernel_launch(void* const* d_in, const int* in_sizes, int n_in,
                              void* d_out, int out_size, void* d_ws, size_t ws_size,
                              hipStream_t stream) {
  (void)in_sizes; (void)n_in; (void)out_size; (void)ws_size;
  const int*   word_ixs  = (const int*)  d_in[0];
  const int*   char_ixs  = (const int*)  d_in[1];
  const int*   char_lens = (const int*)  d_in[2];
  const float* word_emb  = (const float*)d_in[3];
  const float* char_emb  = (const float*)d_in[4];
  const float* c_wih     = (const float*)d_in[5];
  const float* c_whh     = (const float*)d_in[6];
  const float* c_bih     = (const float*)d_in[7];
  const float* c_bhh     = (const float*)d_in[8];
  const float* w_wih     = (const float*)d_in[9];
  const float* w_whh     = (const float*)d_in[10];
  const float* w_bih     = (const float*)d_in[11];
  const float* w_bhh     = (const float*)d_in[12];
  const float* out_w     = (const float*)d_in[13];
  const float* out_b     = (const float*)d_in[14];
  float* out = (float*)d_out;

  char* ws = (char*)d_ws;
  size_t off = 0;
  auto carve = [&](size_t bytes) {
    size_t o = off;
    off = (off + bytes + 255) & ~(size_t)255;
    return o;
  };
  unsigned int* wcf = (unsigned int*)(ws + carve((size_t)NT1 * KC1 * 32 * 8 * 4)); // 64 KB
  unsigned int* wif = (unsigned int*)(ws + carve((size_t)NT2 * KC2 * 32 * 8 * 4)); // 384 KB
  unsigned int* owf = (unsigned int*)(ws + carve((size_t)NT4 * KC4 * 32 * 8 * 4)); // 64 KB
  float* cb     = (float*)(ws + carve((size_t)ZC * 4));
  float* wb     = (float*)(ws + carve((size_t)ZW * 4));
  float* h_char = (float*)(ws + carve((size_t)SEQ * CRv * 4));   // 2 MB
  float* XZ     = (float*)(ws + carve((size_t)SEQ * ZW * 4));    // 32 MB
  float* hs     = (float*)(ws + carve((size_t)SEQ * HDIM * 4));  // 8 MB

  const int prepN = NT1 * KC1 * 32 * 8 + NT2 * KC2 * 32 * 8 + NT4 * KC4 * 32 * 8 + ZC + ZW;
  prep_kernel<<<(prepN + 255) / 256, 256, 0, stream>>>(
      c_wih, c_whh, c_bih, c_bhh, w_wih, w_bih, w_bhh, out_w, wcf, wif, owf, cb, wb);

  // 512 waves of 16 words each = 16384 threads
  char_lstm_kernel<<<128, 128, 0, stream>>>(char_ixs, char_lens, char_emb, wcf, cb, h_char);
  word_inproj_kernel<<<128, 128, 0, stream>>>(word_ixs, word_emb, h_char, wif, wb, XZ);
  word_lstm_kernel<<<1, 1024, 0, stream>>>(XZ, w_whh, hs);
  out_softmax_kernel<<<128, 128, 0, stream>>>(hs, owf, out_b, out);
}
